// BipartiteGCN_4887672783078
// MI455X (gfx1250) — matrix operand order repaired and checked by reading, not compiled, not run on hardware
//
#include <hip/hip_runtime.h>
#include <hip/hip_bf16.h>

typedef _Float16 h16;
typedef __attribute__((ext_vector_type(16))) _Float16 v16h;
typedef __attribute__((ext_vector_type(8)))  _Float16 v8h;
typedef __attribute__((ext_vector_type(8)))  float    v8f;

#define EMB 64
#define SLOPE 0.01f
#define EPSF 1e-5f

__device__ __forceinline__ float lrelu(float x) { return x >= 0.0f ? x : SLOPE * x; }

// ---- WMMA fragment helpers (CDNA5 ISA 7.12.2 layouts) ------------------------------------------
// A fragment: 16x32 f16 tile, row-major in LDS (leading dim ld), K-slice at kbase.
// lanes 0-15: M=lane, K = {0..7,16..23}+kbase ; lanes 16-31: M=lane-16, K = {8..15,24..31}+kbase.
// Both 8-half runs are contiguous -> two ds_load_b128.
__device__ __forceinline__ v16h load_a_frag(const h16* a, int ld, int kbase, int lane) {
  const h16* p = a + (size_t)(lane & 15) * ld + kbase + ((lane < 16) ? 0 : 8);
  v8h lo = *(const v8h*)p;
  v8h hi = *(const v8h*)(p + 16);
  return __builtin_shufflevector(lo, hi, 0, 1, 2, 3, 4, 5, 6, 7, 8, 9, 10, 11, 12, 13, 14, 15);
}

// B fragments are staged PRE-SWIZZLED: fragment f, lane l owns 16 contiguous halfs.
__device__ __forceinline__ v16h load_b_swz(const h16* sWs, int f, int lane) {
  return *(const v16h*)(sWs + (((f * 32) + lane) << 4));
}

// Swizzle W[K][64] (f32, row-major) into per-fragment per-lane contiguous f16 layout.
// Fragment f = (k>>5)*4 + (n>>4); lane = ((k>>3)&1)*16 + (n&15); elem i = (k&7) | ((k>>1)&8).
__device__ __forceinline__ void stage_b_swz(const float* __restrict__ W, h16* __restrict__ sWs,
                                            int K, int tid) {
  int total = K * EMB;
  for (int idx = tid; idx < total; idx += 256) {
    int k = idx >> 6, n = idx & 63;
    int f = (k >> 5) * 4 + (n >> 4);
    int lane = ((k >> 3) & 1) * 16 + (n & 15);
    sWs[(((f * 32) + lane) << 4) + ((k & 7) | ((k >> 1) & 8))] = (h16)W[idx];
  }
}

__device__ __forceinline__ v8f wmma_f16(v16h a, v16h b, v8f c) {
  return __builtin_amdgcn_wmma_f32_16x16x32_f16(false, a, false, b, (short)0, c, false, false);
}

// ---------------- Embed: LN(F) -> Linear(F,64) -> LReLU -> Linear(64,64) -> LReLU [+ pos_emb] ---
template<int F, bool POS>
__global__ __launch_bounds__(256)
void embed_kernel(const float* __restrict__ x,
                  const float* __restrict__ g, const float* __restrict__ b,
                  const float* __restrict__ W1, const float* __restrict__ b1,
                  const float* __restrict__ W2, const float* __restrict__ b2,
                  const float* __restrict__ pos_emb, const int* __restrict__ flags,
                  float* __restrict__ out, int N) {
  __shared__ __align__(32) h16 sW2[EMB * EMB];
  __shared__ __align__(16) h16 sH[64 * EMB];
  __shared__ float sXn[64 * F];

  int tid = threadIdx.x;
  int rowBase = blockIdx.x * 64;

  stage_b_swz(W2, sW2, EMB, tid);

  if (tid < 64) {
    int row = rowBase + tid;
    if (row < N) {
      const float* xr = x + (size_t)row * F;
      float m = 0.0f, v = 0.0f;
#pragma unroll
      for (int f = 0; f < F; ++f) m += xr[f];
      m /= (float)F;
#pragma unroll
      for (int f = 0; f < F; ++f) { float d = xr[f] - m; v += d * d; }
      v /= (float)F;
      float inv = rsqrtf(v + EPSF);
#pragma unroll
      for (int f = 0; f < F; ++f) sXn[tid * F + f] = (xr[f] - m) * inv * g[f] + b[f];
    } else {
#pragma unroll
      for (int f = 0; f < F; ++f) sXn[tid * F + f] = 0.0f;
    }
  }
  __syncthreads();

  // Linear1 + LReLU (F tiny: VALU), stage f16 activations for WMMA
#pragma unroll
  for (int i = 0; i < 16; ++i) {
    int idx = tid * 16 + i;
    int node = idx >> 6, j = idx & 63;
    float acc = b1[j];
#pragma unroll
    for (int f = 0; f < F; ++f) acc += sXn[node * F + f] * W1[f * EMB + j];
    sH[idx] = (h16)lrelu(acc);
  }
  __syncthreads();

  int wv = tid >> 5, lane = tid & 31;
  bool full = (rowBase + 64) <= N;
#pragma unroll
  for (int t = 0; t < 2; ++t) {
    int tile = wv + t * 8;
    int mt = tile >> 2, nt = tile & 3;
    v16h a0 = load_a_frag(sH + mt * 16 * EMB, EMB, 0, lane);
    v16h a1 = load_a_frag(sH + mt * 16 * EMB, EMB, 32, lane);
    v8f c = {};
    c = wmma_f16(a0, load_b_swz(sW2, nt, lane), c);
    c = wmma_f16(a1, load_b_swz(sW2, 4 + nt, lane), c);
    int col = nt * 16 + (lane & 15);
    float bc = b2[col];
    int row0 = rowBase + mt * 16 + ((lane < 16) ? 0 : 8);
    if (full) {
      float* py = out + (size_t)row0 * EMB + col;
      const int* pf = POS ? (flags + row0) : nullptr;
#pragma unroll
      for (int r = 0; r < 8; ++r) {
        float val = lrelu(c[r] + bc);
        if (POS) val += pos_emb[(size_t)pf[r] * EMB + col];
        py[(size_t)r * EMB] = val;
      }
    } else {
#pragma unroll
      for (int r = 0; r < 8; ++r) {
        int grow = row0 + r;
        if (grow < N) {
          float val = lrelu(c[r] + bc);
          if (POS) val += pos_emb[(size_t)flags[grow] * EMB + col];
          out[(size_t)grow * EMB + col] = val;
        }
      }
    }
  }
}

// ---------------- Generic 64x64 Linear: Y = act(X @ W [+ b]) ------------------------------------
template<bool HAS_BIAS, bool ACT>
__global__ __launch_bounds__(256)
void lin64_kernel(const float* __restrict__ X, const float* __restrict__ W,
                  const float* __restrict__ bias, float* __restrict__ Y, int N) {
  __shared__ __align__(32) h16 sW[EMB * EMB];
  __shared__ __align__(16) h16 sX[64 * EMB];
  int tid = threadIdx.x;
  int rowBase = blockIdx.x * 64;

  stage_b_swz(W, sW, EMB, tid);
#pragma unroll
  for (int i = 0; i < 16; ++i) {
    int idx = tid * 16 + i;
    int row = rowBase + (idx >> 6);
    sX[idx] = (row < N) ? (h16)X[(size_t)row * EMB + (idx & 63)] : (h16)0.0f;
  }
  __syncthreads();

  int wv = tid >> 5, lane = tid & 31;
  bool full = (rowBase + 64) <= N;
#pragma unroll
  for (int t = 0; t < 2; ++t) {
    int tile = wv + t * 8;
    int mt = tile >> 2, nt = tile & 3;
    v16h a0 = load_a_frag(sX + mt * 16 * EMB, EMB, 0, lane);
    v16h a1 = load_a_frag(sX + mt * 16 * EMB, EMB, 32, lane);
    v8f c = {};
    c = wmma_f16(a0, load_b_swz(sW, nt, lane), c);
    c = wmma_f16(a1, load_b_swz(sW, 4 + nt, lane), c);
    int col = nt * 16 + (lane & 15);
    float bc = HAS_BIAS ? bias[col] : 0.0f;
    int row0 = rowBase + mt * 16 + ((lane < 16) ? 0 : 8);
    if (full) {
      float* py = Y + (size_t)row0 * EMB + col;
#pragma unroll
      for (int r = 0; r < 8; ++r) {
        float val = c[r] + bc;
        if (ACT) val = lrelu(val);
        py[(size_t)r * EMB] = val;
      }
    } else {
#pragma unroll
      for (int r = 0; r < 8; ++r) {
        int grow = row0 + r;
        if (grow < N) {
          float val = c[r] + bc;
          if (ACT) val = lrelu(val);
          Y[(size_t)grow * EMB + col] = val;
        }
      }
    }
  }
}

// ---------------- Edge message kernel: hot loop over 2M edges -----------------------------------
// msg = LReLU(LN(lj[src] + ri[dst])) @ Wf + bf ; atomic scatter-add into agg[dst]; count edges.
__global__ __launch_bounds__(256)
void edge_msg_kernel(const int* __restrict__ esrc, const int* __restrict__ edst,
                     const float* __restrict__ lj, const float* __restrict__ ri,
                     const float* __restrict__ g_ln, const float* __restrict__ b_ln,
                     const float* __restrict__ Wf, const float* __restrict__ bf,
                     float* __restrict__ agg, float* __restrict__ cnt, int nE) {
  __shared__ __align__(32) h16 sWf[EMB * EMB];
  __shared__ __align__(16) float sMsg[8][16][EMB];
  __shared__ __align__(16) h16 sA[8][16 * EMB];
  __shared__ int sDst[8][16];

  int tid = threadIdx.x;
  int wv = tid >> 5, lane = tid & 31;
  int base = (blockIdx.x * 8 + wv) * 16;

  // Load edge indices early; prefetch the gather rows while the wave stages weights.
  int er = lane >> 1;
  int e = base + er;
  int c0 = (lane & 1) * 32;
  int src = 0, dst = 0;
  bool eok = (e < nE);
  if (eok) {
    src = esrc[e];
    dst = edst[e];
    __builtin_prefetch(lj + (size_t)src * EMB + c0, 0, 0);
    __builtin_prefetch(ri + (size_t)dst * EMB + c0, 0, 0);
  }
  if ((lane & 1) == 0) sDst[wv][er] = dst;

  stage_b_swz(Wf, sWf, EMB, tid);

  // Gather lj[src] + ri[dst]: two lanes per edge, 8 x float4 each.
  {
    float4* pm = (float4*)(&sMsg[wv][er][c0]);
    if (eok) {
      const float4* pl = (const float4*)(lj + (size_t)src * EMB + c0);
      const float4* pr = (const float4*)(ri + (size_t)dst * EMB + c0);
#pragma unroll
      for (int i = 0; i < 8; ++i) {
        float4 a = pl[i], b4 = pr[i];
        float4 o; o.x = a.x + b4.x; o.y = a.y + b4.y; o.z = a.z + b4.z; o.w = a.w + b4.w;
        pm[i] = o;
      }
    } else {
      float4 z = {0.f, 0.f, 0.f, 0.f};
#pragma unroll
      for (int i = 0; i < 8; ++i) pm[i] = z;
    }
  }
  __syncthreads();

  // Per-edge LayerNorm + LReLU across all 32 lanes (2 lanes per row, shfl_xor reduce).
  {
    int row = lane & 15;
    int half = lane >> 4;
    const float* mrow = sMsg[wv][row];
    float s = 0.0f, s2 = 0.0f;
#pragma unroll
    for (int j = 0; j < 32; ++j) {
      float xv = mrow[half * 32 + j];
      s += xv; s2 += xv * xv;
    }
    s  += __shfl_xor(s, 16, 32);
    s2 += __shfl_xor(s2, 16, 32);
    float m = s * (1.0f / EMB);
    float inv = rsqrtf(s2 * (1.0f / EMB) - m * m + EPSF);
#pragma unroll
    for (int j = 0; j < 32; ++j) {
      int jj = half * 32 + j;
      sA[wv][row * EMB + jj] = (h16)lrelu((mrow[jj] - m) * inv * g_ln[jj] + b_ln[jj]);
    }
    if (half == 0) {
      int ee = base + row;
      if (ee < nE) atomicAdd(&cnt[sDst[wv][row]], 1.0f);
    }
  }
  __syncthreads();

  // msg @ Wf + bf, scatter into agg[dst] via global f32 atomics.
  // Precompute 8 per-lane base pointers once; the nt loop only changes the
  // address by nt*16 floats = 64 B, which folds into the atomic's IOFFSET.
  v16h a0 = load_a_frag(sA[wv], EMB, 0, lane);
  v16h a1 = load_a_frag(sA[wv], EMB, 32, lane);
  float* pr[8];
#pragma unroll
  for (int r = 0; r < 8; ++r)
    pr[r] = agg + (size_t)sDst[wv][(lane < 16) ? r : (8 + r)] * EMB + (lane & 15);
  bool full = (base + 16) <= nE;

#pragma unroll
  for (int nt = 0; nt < 4; ++nt) {
    v8f c = {};
    c = wmma_f16(a0, load_b_swz(sWf, nt, lane), c);
    c = wmma_f16(a1, load_b_swz(sWf, 4 + nt, lane), c);
    float bc = bf[nt * 16 + (lane & 15)];
    if (full) {
#pragma unroll
      for (int r = 0; r < 8; ++r)
        atomicAdd(pr[r] + nt * 16, c[r] + bc);
    } else {
#pragma unroll
      for (int r = 0; r < 8; ++r) {
        int ee = base + ((lane < 16) ? r : (8 + r));
        if (ee < nE)
          atomicAdd(pr[r] + nt * 16, c[r] + bc);
      }
    }
  }
}

// ---------------- Post-conv: LN(agg/cnt) ++ right -> Linear(128,64)+LReLU -> Linear(64,64) ------
__global__ __launch_bounds__(256)
void post_conv_kernel(const float* __restrict__ agg, const float* __restrict__ cnt,
                      const float* __restrict__ g_post, const float* __restrict__ b_post,
                      const float* __restrict__ right,
                      const float* __restrict__ Wo1, const float* __restrict__ bo1,
                      const float* __restrict__ Wo2, const float* __restrict__ bo2,
                      float* __restrict__ out, int N) {
  __shared__ __align__(32) h16 sW1[128 * EMB];
  __shared__ __align__(32) h16 sW2[EMB * EMB];
  __shared__ __align__(16) h16 sH[64 * 128];
  __shared__ __align__(16) h16 sH2[64 * EMB];
  __shared__ float sM[64], sInv[64], sC[64];

  int tid = threadIdx.x;
  int rowBase = blockIdx.x * 64;
  stage_b_swz(Wo1, sW1, 128, tid);
  stage_b_swz(Wo2, sW2, EMB, tid);

  if (tid < 64) {
    int row = rowBase + tid;
    if (row < N) {
      float rc = 1.0f / fmaxf(cnt[row], 1.0f);
      const float* ar = agg + (size_t)row * EMB;
      float m = 0.0f, v = 0.0f;
#pragma unroll
      for (int j = 0; j < EMB; ++j) m += ar[j] * rc;
      m *= (1.0f / EMB);
#pragma unroll
      for (int j = 0; j < EMB; ++j) { float d = ar[j] * rc - m; v += d * d; }
      v *= (1.0f / EMB);
      sM[tid] = m; sInv[tid] = rsqrtf(v + EPSF); sC[tid] = rc;
    } else { sM[tid] = 0.0f; sInv[tid] = 0.0f; sC[tid] = 0.0f; }
  }
  __syncthreads();

  // Build concat [LN(agg/cnt) | right] as f16, 64 rows x 128 cols.
#pragma unroll
  for (int i = 0; i < 32; ++i) {
    int idx = tid * 32 + i;
    int row = idx >> 7, col = idx & 127;
    int grow = rowBase + row;
    float val = 0.0f;
    if (grow < N) {
      if (col < EMB)
        val = (agg[(size_t)grow * EMB + col] * sC[row] - sM[row]) * sInv[row] * g_post[col] + b_post[col];
      else
        val = right[(size_t)grow * EMB + (col - EMB)];
    }
    sH[idx] = (h16)val;
  }
  __syncthreads();

  int wv = tid >> 5, lane = tid & 31;
#pragma unroll
  for (int t = 0; t < 2; ++t) {
    int tile = wv + t * 8;
    int mt = tile >> 2, nt = tile & 3;
    v8f c = {};
#pragma unroll
    for (int kc = 0; kc < 4; ++kc) {
      v16h a = load_a_frag(sH + mt * 16 * 128, 128, kc * 32, lane);
      c = wmma_f16(a, load_b_swz(sW1, kc * 4 + nt, lane), c);
    }
    int col = nt * 16 + (lane & 15);
    float bc = bo1[col];
    h16* ph = sH2 + (mt * 16 + ((lane < 16) ? 0 : 8)) * EMB + col;
#pragma unroll
    for (int r = 0; r < 8; ++r)
      ph[r * EMB] = (h16)lrelu(c[r] + bc);
  }
  __syncthreads();

  bool full = (rowBase + 64) <= N;
#pragma unroll
  for (int t = 0; t < 2; ++t) {
    int tile = wv + t * 8;
    int mt = tile >> 2, nt = tile & 3;
    v16h a0 = load_a_frag(sH2 + mt * 16 * EMB, EMB, 0, lane);
    v16h a1 = load_a_frag(sH2 + mt * 16 * EMB, EMB, 32, lane);
    v8f c = {};
    c = wmma_f16(a0, load_b_swz(sW2, nt, lane), c);
    c = wmma_f16(a1, load_b_swz(sW2, 4 + nt, lane), c);
    int col = nt * 16 + (lane & 15);
    float bc = bo2[col];
    int row0 = rowBase + mt * 16 + ((lane < 16) ? 0 : 8);
    if (full) {
      float* py = out + (size_t)row0 * EMB + col;
#pragma unroll
      for (int r = 0; r < 8; ++r)
        py[(size_t)r * EMB] = c[r] + bc;
    } else {
#pragma unroll
      for (int r = 0; r < 8; ++r) {
        int grow = row0 + r;
        if (grow < N) out[(size_t)grow * EMB + col] = c[r] + bc;
      }
    }
  }
}

// ---------------- Small helpers ------------------------------------------------------------------
__global__ void set_flags_kernel(const int* __restrict__ cand, int* __restrict__ flags, int nc) {
  int i = blockIdx.x * blockDim.x + threadIdx.x;
  if (i < nc) flags[cand[i]] = 1;
}

__global__ void head_out_kernel(const float* __restrict__ h, const float* __restrict__ W2,
                                const float* __restrict__ b2, const int* __restrict__ cand,
                                float* __restrict__ out, int nc) {
  int i = blockIdx.x * blockDim.x + threadIdx.x;
  if (i < nc) {
    int c = cand[i];
    const float* hr = h + (size_t)c * EMB;
    float acc = b2[0];
#pragma unroll
    for (int j = 0; j < EMB; ++j) acc += hr[j] * W2[j];
    out[i] = acc;
  }
}

// ---------------- Driver -------------------------------------------------------------------------
extern "C" void kernel_launch(void* const* d_in, const int* in_sizes, int n_in,
                              void* d_out, int out_size, void* d_ws, size_t ws_size,
                              hipStream_t stream) {
  (void)n_in; (void)out_size; (void)ws_size;
  const float* row_features = (const float*)d_in[0];
  // d_in[1] = edge_attr: unused by the reference forward pass
  const float* var_features = (const float*)d_in[2];
  const float* ce_g  = (const float*)d_in[3];
  const float* ce_b  = (const float*)d_in[4];
  const float* ce_W1 = (const float*)d_in[5];
  const float* ce_b1 = (const float*)d_in[6];
  const float* ce_W2 = (const float*)d_in[7];
  const float* ce_b2 = (const float*)d_in[8];
  const float* ve_g  = (const float*)d_in[9];
  const float* ve_b  = (const float*)d_in[10];
  const float* ve_W1 = (const float*)d_in[11];
  const float* ve_b1 = (const float*)d_in[12];
  const float* ve_W2 = (const float*)d_in[13];
  const float* ve_b2 = (const float*)d_in[14];
  const float* pos_emb = (const float*)d_in[15];
  // conv param order: 0 Wl,1 bl,2 Wr,3 g_ln,4 b_ln,5 Wf,6 bf,7 g_post,8 b_post,9 Wo1,10 bo1,11 Wo2,12 bo2
  const float* v2c[13]; for (int i = 0; i < 13; ++i) v2c[i] = (const float*)d_in[16 + i];
  const float* c2v[13]; for (int i = 0; i < 13; ++i) c2v[i] = (const float*)d_in[29 + i];
  const float* h_W1 = (const float*)d_in[42];
  const float* h_b1 = (const float*)d_in[43];
  const float* h_W2 = (const float*)d_in[44];
  const float* h_b2 = (const float*)d_in[45];
  const int* eidx = (const int*)d_in[46];
  const int* cand = (const int*)d_in[47];

  const int NC = in_sizes[0] / 5;
  const int NV = in_sizes[2] / 19;
  const int NE = in_sizes[46] / 2;
  const int NCAND = in_sizes[47];
  const int* edge_cons = eidx;          // edge_index[0]
  const int* edge_var  = eidx + NE;     // edge_index[1]

  float* ws = (float*)d_ws;
  size_t R = (size_t)NV * EMB;
  float* bufA = ws;            // consE -> lj2
  float* bufB = ws + R;        // varE (kept through both convs)
  float* bufC = ws + 2 * R;    // lj(v2c) -> ri2(c2v) -> var2
  float* bufD = ws + 3 * R;    // ri(v2c) -> cons2 -> head hidden
  float* bufE = ws + 4 * R;    // agg (scatter target)
  float* bufCnt = ws + 5 * R;  // in-degree counts
  int* flags = (int*)(bufCnt + NV);

  dim3 blk(256);
  int gCons = (NC + 63) / 64, gVar = (NV + 63) / 64;
  int eTiles = (NE + 15) / 16;
  int gEdge = (eTiles + 7) / 8;

  // candidate one-hot flags
  hipMemsetAsync(flags, 0, (size_t)NV * sizeof(int), stream);
  set_flags_kernel<<<(NCAND + 255) / 256, blk, 0, stream>>>(cand, flags, NCAND);

  // node embeddings
  embed_kernel<5, false><<<gCons, blk, 0, stream>>>(row_features, ce_g, ce_b, ce_W1, ce_b1,
                                                    ce_W2, ce_b2, nullptr, nullptr, bufA, NC);
  embed_kernel<19, true><<<gVar, blk, 0, stream>>>(var_features, ve_g, ve_b, ve_W1, ve_b1,
                                                   ve_W2, ve_b2, pos_emb, flags, bufB, NV);

  // ---- v2c conv (src = var, dst = cons) ----
  lin64_kernel<true,  false><<<gVar,  blk, 0, stream>>>(bufB, v2c[0], v2c[1], bufC, NV);  // lj
  lin64_kernel<false, false><<<gCons, blk, 0, stream>>>(bufA, v2c[2], nullptr, bufD, NC); // ri
  hipMemsetAsync(bufE, 0, (size_t)NC * EMB * sizeof(float), stream);
  hipMemsetAsync(bufCnt, 0, (size_t)NC * sizeof(float), stream);
  edge_msg_kernel<<<gEdge, blk, 0, stream>>>(edge_var, edge_cons, bufC, bufD,
                                             v2c[3], v2c[4], v2c[5], v2c[6], bufE, bufCnt, NE);
  post_conv_kernel<<<gCons, blk, 0, stream>>>(bufE, bufCnt, v2c[7], v2c[8], bufA,
                                              v2c[9], v2c[10], v2c[11], v2c[12], bufD, NC); // cons2

  // ---- c2v conv (src = cons, dst = var) ----
  lin64_kernel<true,  false><<<gCons, blk, 0, stream>>>(bufD, c2v[0], c2v[1], bufA, NC);  // lj2
  lin64_kernel<false, false><<<gVar,  blk, 0, stream>>>(bufB, c2v[2], nullptr, bufC, NV); // ri2
  hipMemsetAsync(bufE, 0, (size_t)NV * EMB * sizeof(float), stream);
  hipMemsetAsync(bufCnt, 0, (size_t)NV * sizeof(float), stream);
  edge_msg_kernel<<<gEdge, blk, 0, stream>>>(edge_cons, edge_var, bufA, bufC,
                                             c2v[3], c2v[4], c2v[5], c2v[6], bufE, bufCnt, NE);
  post_conv_kernel<<<gVar, blk, 0, stream>>>(bufE, bufCnt, c2v[7], c2v[8], bufB,
                                             c2v[9], c2v[10], c2v[11], c2v[12], bufC, NV); // var2

  // ---- head ----
  lin64_kernel<true, true><<<gVar, blk, 0, stream>>>(bufC, h_W1, h_b1, bufD, NV);
  head_out_kernel<<<(NCAND + 255) / 256, blk, 0, stream>>>(bufD, h_W2, h_b2, cand,
                                                           (float*)d_out, NCAND);
}
